// RNNDecoder_39359080300577
// MI455X (gfx1250) — matrix operand order repaired and checked by reading, not compiled, not run on hardware
//
#include <hip/hip_runtime.h>

// ---------------------------------------------------------------------------
// LSTM + attention decoder, B=8192 S=128 E=128 H=128 F=3 T=10 (MI455X/gfx1250).
//  * Workgroup owns 8 batch rows for all 10 steps; its 8x128x128 encoder tile
//    is pinned in LDS as f16 (256 KB of 320 KB/WGP) -> encoder_outs (512 MB)
//    crosses HBM exactly once (~25us floor) instead of 10-20x.
//  * Shared-weight GEMMs (gates/q/attn-out) use v_wmma_f32_16x16x32_f16 with
//    f32 accumulate; the compiler hoists the (loop-invariant) weight fragments
//    into VGPRs for the whole time loop.
//  * Per-row attention GEMVs run on VALU from LDS with v_dot2_f32_f16 (both
//    the scores GEMV and the ctx reduction), f32 accumulation throughout.
//  * Nonlinearities are v_exp_f32/v_rcp_f32 trans ops (co-issue with WMMA).
// ---------------------------------------------------------------------------

typedef _Float16 f16_t;
typedef __attribute__((ext_vector_type(16))) _Float16 v16h;
typedef __attribute__((ext_vector_type(2)))  _Float16 h2_t;
typedef __attribute__((ext_vector_type(8)))  float    v8f;

static constexpr int Bt = 8192, S = 128, E = 128, H = 128, F = 3, T = 10;
static constexpr int NB = 8;           // batch rows per workgroup
static constexpr int NT = 512;         // 16 wave32s
static constexpr int G4 = 4 * H;       // 512 gate columns
static constexpr int CW = E + H;       // 256, ctxh row width

// LDS layout (bytes). Total 312,976 B < 327,680 (320 KB/WGP).
static constexpr int OFF_GATES = 0;                      // f32 [NB][512]
static constexpr int OFF_C     = OFF_GATES + NB * G4 * 4;// f32 [NB][H]
static constexpr int OFF_ATTN  = OFF_C     + NB * H * 4; // f32 [NB][S] scores
static constexpr int OFF_HID   = OFF_ATTN  + NB * S * 4; // f32 [NB][H]
static constexpr int OFF_INP   = OFF_HID   + NB * H * 4; // f32 [NB][4]
static constexpr int OFF_Q     = OFF_INP   + NB * 4 * 4; // f16 [NB][E]
static constexpr int OFF_ATTNH = OFF_Q     + NB * E * 2; // f16 [NB][S] softmax
static constexpr int OFF_CTXH  = OFF_ATTNH + NB * S * 2; // f16 [16][256]
static constexpr int OFF_ENC   = OFF_CTXH  + 16 * CW * 2;// f16 [NB][S][E]
static constexpr int OFF_BIAS  = OFF_ENC   + NB * S * E * 2; // f32 [512]
static constexpr int OFF_WIH   = OFF_BIAS  + G4 * 4;     // f32 [512][3]
static constexpr int OFF_WOUT  = OFF_WIH   + G4 * 3 * 4; // f32 [3][128]
static constexpr int OFF_BOUT  = OFF_WOUT  + F * H * 4;  // f32 [4]
static constexpr int SMEM_BYTES = OFF_BOUT + 4 * 4;

__device__ inline float rcp_fast(float x) { return __builtin_amdgcn_rcpf(x); }
__device__ inline float sigmoid_fast(float x) { return rcp_fast(1.0f + __expf(-x)); }
__device__ inline float tanh_fast(float x) {
  // 1 - 2/(e^{2x}+1): exp->inf => 1.0, exp->0 => -1.0. v_exp_f32 + v_rcp_f32.
  return 1.0f - 2.0f * rcp_fast(__expf(2.0f * x) + 1.0f);
}

// A fragment (16x32 f16) from LDS: lane m holds K{koff..koff+15},
// koff = (lane>=16)*16.
__device__ inline v16h load_a_frag(const f16_t* ctxh, int colbase, int kb, int lane) {
  const int m    = lane & 15;
  const int koff = (lane >> 4) << 4;
  const f16_t* p = ctxh + m * CW + colbase + kb + koff;
  v16h a;
#pragma unroll
  for (int i = 0; i < 8; ++i) a[i] = p[i];
#pragma unroll
  for (int i = 0; i < 8; ++i) a[8 + i] = p[8 + i];
  return a;
}

// B fragment (32x16 f16) from row-major f32 weights W[N][K]; B[k][n]=W[n][k].
// Same contiguous k order as A: lanes 0-15 hold K 0..15 of column n, lanes 16-31 hold K 16..31.
__device__ inline v16h load_w_frag(const float* __restrict__ W, int Kdim,
                                   int n, int kb, int lane) {
  const int kstart = kb + ((lane >> 4) << 4);
  const float* p = W + (size_t)n * Kdim + kstart;
  float4 w0 = *(const float4*)(p);
  float4 w1 = *(const float4*)(p + 4);
  float4 w2 = *(const float4*)(p + 8);
  float4 w3 = *(const float4*)(p + 12);
  v16h b;
  b[0]  = (f16_t)w0.x; b[1]  = (f16_t)w0.y; b[2]  = (f16_t)w0.z; b[3]  = (f16_t)w0.w;
  b[4]  = (f16_t)w1.x; b[5]  = (f16_t)w1.y; b[6]  = (f16_t)w1.z; b[7]  = (f16_t)w1.w;
  b[8]  = (f16_t)w2.x; b[9]  = (f16_t)w2.y; b[10] = (f16_t)w2.z; b[11] = (f16_t)w2.w;
  b[12] = (f16_t)w3.x; b[13] = (f16_t)w3.y; b[14] = (f16_t)w3.z; b[15] = (f16_t)w3.w;
  return b;
}

__device__ inline float dot2_f16(h2_t a, h2_t b, float c) {
#if __has_builtin(__builtin_amdgcn_fdot2)
  return __builtin_amdgcn_fdot2(a, b, c, false);   // v_dot2_f32_f16
#else
  return c + (float)a[0] * (float)b[0] + (float)a[1] * (float)b[1];
#endif
}

__global__ void __launch_bounds__(NT, 1)
rnn_attn_decoder(const float* __restrict__ start_decode,
                 const float* __restrict__ encoder_outs,
                 const float* __restrict__ encoder_hiddens,
                 const float* __restrict__ encoder_cells,
                 const float* __restrict__ W_ih,
                 const float* __restrict__ W_hh,
                 const float* __restrict__ b_ih,
                 const float* __restrict__ b_hh,
                 const float* __restrict__ W_attn_in,
                 const float* __restrict__ W_attn_out,
                 const float* __restrict__ W_out,
                 const float* __restrict__ b_out,
                 float* __restrict__ ys_out,
                 float* __restrict__ hids_out) {
  extern __shared__ __align__(16) char smem[];
  float* gatesL = (float*)(smem + OFF_GATES);
  float* cL     = (float*)(smem + OFF_C);
  float* attnL  = (float*)(smem + OFF_ATTN);   // f32 raw scores
  float* hidL   = (float*)(smem + OFF_HID);
  float* inpL   = (float*)(smem + OFF_INP);
  f16_t* qL     = (f16_t*)(smem + OFF_Q);
  f16_t* attnHL = (f16_t*)(smem + OFF_ATTNH);  // f16 softmax weights
  f16_t* ctxhL  = (f16_t*)(smem + OFF_CTXH);   // [16][256]: [ctx | h], rows 8..15 zero
  f16_t* encL   = (f16_t*)(smem + OFF_ENC);    // [NB][S][E]
  float* biasL  = (float*)(smem + OFF_BIAS);   // b_ih + b_hh
  float* wihL   = (float*)(smem + OFF_WIH);    // W_ih (512x3)
  float* woutL  = (float*)(smem + OFF_WOUT);   // W_out (3x128)
  float* boutL  = (float*)(smem + OFF_BOUT);   // b_out (3)

  const int tid  = threadIdx.x;
  const int lane = tid & 31;
  const int wave = tid >> 5;
  const int b0   = blockIdx.x * NB;

  // ---------------- preload (once; encoder stays LDS-resident for all T) ----
  {
    const float* eg = encoder_outs + (size_t)b0 * S * E;
#pragma unroll 4
    for (int i = 0; i < (NB * S * E) / (NT * 4); ++i) {   // 64 iters
      int base = (i * NT + tid) * 4;
      float4 v = *(const float4*)(eg + base);
      f16_t* p = encL + base;
      p[0] = (f16_t)v.x; p[1] = (f16_t)v.y; p[2] = (f16_t)v.z; p[3] = (f16_t)v.w;
    }
#pragma unroll
    for (int k = 0; k < (8 * CW) / NT; ++k)               // zero rows 8..15
      ctxhL[8 * CW + k * NT + tid] = (f16_t)0.0f;
#pragma unroll
    for (int k = 0; k < (NB * H) / NT; ++k) {             // h0, c0
      int idx = k * NT + tid; int b = idx >> 7; int j = idx & 127;
      ctxhL[b * CW + E + j] = (f16_t)encoder_hiddens[(size_t)(b0 + b) * H + j];
      cL[idx] = encoder_cells[(size_t)(b0 + b) * H + j];
    }
    if (tid < G4) biasL[tid] = b_ih[tid] + b_hh[tid];
#pragma unroll
    for (int k = 0; k < 3; ++k) {                          // W_ih: 1536 f32
      int idx = k * NT + tid;
      wihL[idx] = W_ih[idx];
    }
    if (tid < F * H) woutL[tid] = W_out[tid];
    if (tid < F)     boutL[tid] = b_out[tid];
    if (tid < NB * F) {
      int b = tid / F, f = tid % F;
      inpL[b * 4 + f] = start_decode[(size_t)(b0 + b) * F + f];
    }
    // Warm L2 with the shared weights (global_prefetch_b8).
    __builtin_prefetch(W_hh       + (tid & 255) * 256, 0, 0);
    __builtin_prefetch(W_attn_out + (tid & 127) * 256, 0, 0);
    __builtin_prefetch(W_attn_in  + (tid & 127) * 128, 0, 0);
  }
  __syncthreads();

  for (int t = 0; t < T; ++t) {
    // (a) gates = (b_ih + b_hh) + inp @ W_ih^T   (K=3, trivial VALU, all LDS)
#pragma unroll
    for (int k = 0; k < (NB * G4) / NT; ++k) {            // 8 per thread
      int p = k * NT + tid; int b = p >> 9; int j = p & 511;
      const float* wr = wihL + j * 3;
      const float* ir = inpL + b * 4;
      gatesL[p] = biasL[j] + ir[0] * wr[0] + ir[1] * wr[1] + ir[2] * wr[2];
    }
    __syncthreads();

    // (b) gates += h @ W_hh^T  --- WMMA, 32 N-tiles over 16 waves
#pragma unroll
    for (int i = 0; i < 2; ++i) {
      const int nt  = wave * 2 + i;
      const int col = nt * 16 + (lane & 15);
      v8f acc;
      if (lane < 16) {
#pragma unroll
        for (int r = 0; r < 8; ++r) acc[r] = gatesL[r * G4 + col];
      } else {
#pragma unroll
        for (int r = 0; r < 8; ++r) acc[r] = 0.0f;
      }
#pragma unroll
      for (int kb = 0; kb < H; kb += 32) {
        v16h a = load_a_frag(ctxhL, E, kb, lane);
        v16h w = load_w_frag(W_hh, H, col, kb, lane);
        acc = __builtin_amdgcn_wmma_f32_16x16x32_f16(false, a, false, w,
                                                     (short)0, acc, false, false);
      }
      if (lane < 16) {
#pragma unroll
        for (int r = 0; r < 8; ++r) gatesL[r * G4 + col] = acc[r];
      }
    }
    __syncthreads();

    // (c) LSTM cell elementwise (f32, fast trans-op nonlinearities)
#pragma unroll
    for (int k = 0; k < (NB * H) / NT; ++k) {
      int idx = k * NT + tid; int b = idx >> 7; int j = idx & 127;
      const float* g = gatesL + b * G4;
      float si = sigmoid_fast(g[j]);
      float sf = sigmoid_fast(g[H + j]);
      float so = sigmoid_fast(g[3 * H + j]);
      float cn = sf * cL[idx] + si * tanh_fast(g[2 * H + j]);
      cL[idx] = cn;
      ctxhL[b * CW + E + j] = (f16_t)(so * tanh_fast(cn));   // h_new
    }
    __syncthreads();

    // (d) q = h_new @ W_attn_in^T  --- WMMA, 8 N-tiles on waves 0..7
    if (wave < 8) {
      const int col = wave * 16 + (lane & 15);
      v8f acc;
#pragma unroll
      for (int r = 0; r < 8; ++r) acc[r] = 0.0f;
#pragma unroll
      for (int kb = 0; kb < H; kb += 32) {
        v16h a = load_a_frag(ctxhL, E, kb, lane);
        v16h w = load_w_frag(W_attn_in, H, col, kb, lane);
        acc = __builtin_amdgcn_wmma_f32_16x16x32_f16(false, a, false, w,
                                                     (short)0, acc, false, false);
      }
      if (lane < 16) {
#pragma unroll
        for (int r = 0; r < 8; ++r) qL[r * E + col] = (f16_t)acc[r];
      }
    }
    __syncthreads();

    // (e) scores[b][s] = enc[b][s][:] . q[b][:]   (v_dot2_f32_f16 GEMV)
#pragma unroll
    for (int k = 0; k < (NB * S) / NT; ++k) {
      int p = k * NT + tid; int b = p >> 7; int s = p & 127;
      const f16_t* er = encL + ((size_t)b * S + s) * E;
      const f16_t* qr = qL + b * E;
      float acc = 0.0f;
#pragma unroll 16
      for (int e = 0; e < E; e += 2) {
        h2_t ev = *(const h2_t*)(er + e);
        h2_t qv = *(const h2_t*)(qr + e);
        acc = dot2_f16(ev, qv, acc);
      }
      attnL[p] = acc;
    }
    __syncthreads();

    // (f) softmax over s, one wave32 per batch row; result stored packed f16
    if (wave < NB) {
      const int b = wave;
      float v[4];
#pragma unroll
      for (int k = 0; k < 4; ++k) v[k] = attnL[b * S + lane + 32 * k];
      float mx = fmaxf(fmaxf(v[0], v[1]), fmaxf(v[2], v[3]));
#pragma unroll
      for (int off = 16; off >= 1; off >>= 1) mx = fmaxf(mx, __shfl_xor(mx, off, 32));
      float sum = 0.0f;
#pragma unroll
      for (int k = 0; k < 4; ++k) { v[k] = __expf(v[k] - mx); sum += v[k]; }
#pragma unroll
      for (int off = 16; off >= 1; off >>= 1) sum += __shfl_xor(sum, off, 32);
      float inv = rcp_fast(sum);
#pragma unroll
      for (int k = 0; k < 4; ++k)
        attnHL[b * S + lane + 32 * k] = (f16_t)(v[k] * inv);
    }
    __syncthreads();

    // (g) ctx[b][e] = sum_s attn[b][s] * enc[b][s][e]  (dot2 over s-pairs)
    {
      const int b  = tid >> 6;
      const int e0 = (tid & 63) * 2;
      const f16_t* eb = encL + (size_t)b * S * E + e0;
      const f16_t* ah = attnHL + b * S;
      float a0 = 0.0f, a1 = 0.0f;
#pragma unroll 8
      for (int s = 0; s < S; s += 2) {
        h2_t ev0 = *(const h2_t*)(eb + (size_t)s * E);        // row s,  e0..e0+1
        h2_t ev1 = *(const h2_t*)(eb + (size_t)(s + 1) * E);  // row s+1,e0..e0+1
        h2_t aw  = *(const h2_t*)(ah + s);                    // attn s, s+1
        h2_t c0; c0[0] = ev0[0]; c0[1] = ev1[0];
        h2_t c1; c1[0] = ev0[1]; c1[1] = ev1[1];
        a0 = dot2_f16(c0, aw, a0);
        a1 = dot2_f16(c1, aw, a1);
      }
      ctxhL[b * CW + e0]     = (f16_t)a0;
      ctxhL[b * CW + e0 + 1] = (f16_t)a1;
    }
    __syncthreads();

    // (h) hid = tanh([ctx|h] @ W_attn_out^T) --- WMMA K=256, waves 0..7
    if (wave < 8) {
      const int col = wave * 16 + (lane & 15);
      v8f acc;
#pragma unroll
      for (int r = 0; r < 8; ++r) acc[r] = 0.0f;
#pragma unroll
      for (int kb = 0; kb < CW; kb += 32) {
        v16h a = load_a_frag(ctxhL, 0, kb, lane);
        v16h w = load_w_frag(W_attn_out, CW, col, kb, lane);
        acc = __builtin_amdgcn_wmma_f32_16x16x32_f16(false, a, false, w,
                                                     (short)0, acc, false, false);
      }
      if (lane < 16) {
#pragma unroll
        for (int r = 0; r < 8; ++r) {
          float hv = tanh_fast(acc[r]);
          hidL[r * H + col] = hv;
          hids_out[((size_t)(b0 + r) * T + t) * H + col] = hv;
        }
      }
    }
    __syncthreads();

    // (i) y = hid @ W_out^T + b_out  (24 dot-128s, all-LDS) -> next inp + ys
    if (tid < NB * F) {
      int b = tid / F, f = tid % F;
      const float* hr = hidL + b * H;
      const float* wr = woutL + f * H;
      float acc = boutL[f];
#pragma unroll 8
      for (int k2 = 0; k2 < H; ++k2) acc += hr[k2] * wr[k2];
      inpL[b * 4 + f] = acc;
      ys_out[((size_t)(b0 + b) * T + t) * F + f] = acc;
    }
    __syncthreads();
  }
}

extern "C" void kernel_launch(void* const* d_in, const int* in_sizes, int n_in,
                              void* d_out, int out_size, void* d_ws, size_t ws_size,
                              hipStream_t stream) {
  const float* start_decode    = (const float*)d_in[0];
  const float* encoder_outs    = (const float*)d_in[1];
  const float* encoder_hiddens = (const float*)d_in[2];
  const float* encoder_cells   = (const float*)d_in[3];
  const float* W_ih            = (const float*)d_in[4];
  const float* W_hh            = (const float*)d_in[5];
  const float* b_ih            = (const float*)d_in[6];
  const float* b_hh            = (const float*)d_in[7];
  const float* W_attn_in       = (const float*)d_in[8];
  const float* W_attn_out      = (const float*)d_in[9];
  const float* W_out           = (const float*)d_in[10];
  const float* b_out           = (const float*)d_in[11];

  float* out  = (float*)d_out;
  float* ys   = out;                              // (B, T, F)
  float* hids = out + (size_t)Bt * T * F;         // (B, T, H)

  (void)hipFuncSetAttribute((const void*)rnn_attn_decoder,
                            hipFuncAttributeMaxDynamicSharedMemorySize, SMEM_BYTES);

  hipLaunchKernelGGL(rnn_attn_decoder, dim3(Bt / NB), dim3(NT), SMEM_BYTES, stream,
                     start_decode, encoder_outs, encoder_hiddens, encoder_cells,
                     W_ih, W_hh, b_ih, b_hh, W_attn_in, W_attn_out, W_out, b_out,
                     ys, hids);
}
